// Physics_Attention_Irregular_Mesh_36842229465720
// MI455X (gfx1250) — matrix operand order repaired and checked by reading, not compiled, not run on hardware
//
#include <hip/hip_runtime.h>
#include <hip/hip_bf16.h>

typedef __attribute__((ext_vector_type(16))) _Float16 v16h;
typedef __attribute__((ext_vector_type(8)))  _Float16 v8h;
typedef __attribute__((ext_vector_type(8)))  float    v8f;
typedef __attribute__((ext_vector_type(4)))  float    v4f;

#define WMMA_F16(A, B, C) \
    __builtin_amdgcn_wmma_f32_16x16x32_f16(false, (A), false, (B), (short)0, (C), false, false)

// ---------------------------------------------------------------------------
// Fragment-order LDS swizzles (wave32, per CDNA5 ISA §7.12.2).
// LDS holds fragments as [tile][lane][half] so each lane reads one contiguous
// 32-byte v16h (wide ds_load_b128 pairs, no scalar gathers).
//   A forward: lane l, half i -> m = l&15, k = 16*(i>>3) + 8*(l>>4) + (i&7)
//   B forward: lane l, half i -> n = l&15, k = 16*(i>>3) + 8*(l>>4) + (i&7)
// ---------------------------------------------------------------------------
__device__ __forceinline__ int a_swizzle(int m, int k) {   // within 16x32 tile (512 halves)
    return ((m + (((k >> 3) & 1) << 4)) << 4) + ((k >> 4) << 3) + (k & 7);
}
__device__ __forceinline__ int b_swizzle(int k, int n) {   // within 32x16 tile (512 halves)
    return ((n + (((k >> 3) & 1) << 4)) << 4) + ((k >> 4) << 3) + (k & 7);
}

// ---------------------------------------------------------------------------
#define BB 4
#define NN 16384
#define DIMM 256
#define HH 8
#define DD 64
#define GG 32
#define INNER 512

// ---------------------------------------------------------------------------
// K1: x @ Wfx and x @ Wx (WMMA), fused slice logits (WMMA) + softmax + norm.
// grid = (512, 8), block = 256 (8 waves). 128-row x 64-col (= one head) tile.
// ---------------------------------------------------------------------------
__global__ void __launch_bounds__(256)
proj_kernel(const float* __restrict__ x,
            const float* __restrict__ Wfx, const float* __restrict__ bfx,
            const float* __restrict__ Wx,  const float* __restrict__ bx,
            const float* __restrict__ Wslice, const float* __restrict__ bslice,
            const float* __restrict__ temp,
            _Float16* __restrict__ fx_out,   // [B,H,N,D] f16
            float* __restrict__ sw_out,      // [B,H,N,G] f32
            float* __restrict__ slice_norm)  // [B,H,G]  f32 (pre-zeroed)
{
    __shared__ alignas(32) _Float16 sA [8 * 512];   // 8 A-frag tiles (128x32)
    __shared__ alignas(32) _Float16 sBf[4 * 512];   // 4 B-frag tiles (32x64)
    __shared__ alignas(32) _Float16 sBx[4 * 512];
    __shared__ alignas(32) _Float16 sXm[16 * 512];  // x_mid as 16 A-frag tiles (128x64)
    __shared__ alignas(32) _Float16 sWs[4 * 512];   // Wslice as 4 B-frag tiles (64x32)
    __shared__ float sLog[128 * 32];
    __shared__ float snorm[32];

    const int row0 = blockIdx.x * 128;
    const int h    = blockIdx.y;
    const int col0 = h * 64;
    const int tid  = threadIdx.x;
    const int wave = tid >> 5;
    const int lane = tid & 31;

    v8f accF[4] = {};
    v8f accX[4] = {};

    const v16h* pA  = (const v16h*)sA;
    const v16h* pBf = (const v16h*)sBf;
    const v16h* pBx = (const v16h*)sBx;

    for (int ks = 0; ks < DIMM; ks += 32) {
        __syncthreads();
        // A tile: thread loads 16 contiguous floats of x, scatters to frag layout
        {
            int r = tid >> 1, kk0 = (tid & 1) << 4;
            const v4f* xp = (const v4f*)(x + (size_t)(row0 + r) * DIMM + ks + kk0);
            float tmp[16];
            *(v4f*)(tmp + 0)  = xp[0];
            *(v4f*)(tmp + 4)  = xp[1];
            *(v4f*)(tmp + 8)  = xp[2];
            *(v4f*)(tmp + 12) = xp[3];
            int mt = r >> 4, m = r & 15;
#pragma unroll
            for (int j = 0; j < 16; ++j)
                sA[mt * 512 + a_swizzle(m, kk0 + j)] = (_Float16)tmp[j];
        }
        // B tiles: thread loads 8 contiguous floats of each weight row
        {
            int kr = tid >> 3, c0 = (tid & 7) << 3;
            const v4f* wf = (const v4f*)(Wfx + (size_t)(ks + kr) * INNER + col0 + c0);
            const v4f* wx = (const v4f*)(Wx  + (size_t)(ks + kr) * INNER + col0 + c0);
            float tf[8], tx[8];
            *(v4f*)(tf + 0) = wf[0]; *(v4f*)(tf + 4) = wf[1];
            *(v4f*)(tx + 0) = wx[0]; *(v4f*)(tx + 4) = wx[1];
#pragma unroll
            for (int j = 0; j < 8; ++j) {
                int c = c0 + j;
                int idx = (c >> 4) * 512 + b_swizzle(kr, c & 15);
                sBf[idx] = (_Float16)tf[j];
                sBx[idx] = (_Float16)tx[j];
            }
        }
        __syncthreads();
        v16h a = pA[wave * 32 + lane];     // one contiguous 32B fragment read
#pragma unroll
        for (int nt = 0; nt < 4; ++nt) {
            accF[nt] = WMMA_F16(a, pBf[nt * 32 + lane], accF[nt]);
            accX[nt] = WMMA_F16(a, pBx[nt * 32 + lane], accX[nt]);
        }
    }

    const int b     = row0 >> 14;
    const int nbase = row0 & (NN - 1);
    const int nlo   = lane & 15;
    const int mhi   = (lane >> 4) << 3;

    // Epilogue: bias + store fx (f16), stash x_mid to LDS in A-fragment layout
#pragma unroll
    for (int nt = 0; nt < 4; ++nt) {
        int d = nt * 16 + nlo;
        float bf  = bfx[col0 + d];
        float bxv = bx [col0 + d];
        int kc = d >> 5, kk = d & 31;
#pragma unroll
        for (int r = 0; r < 8; ++r) {
            int m  = mhi + r;
            int n  = nbase + wave * 16 + m;
            fx_out[((size_t)(b * HH + h) * NN + n) * DD + d] =
                (_Float16)(accF[nt][r] + bf);
            sXm[(wave * 2 + kc) * 512 + a_swizzle(m, kk)] =
                (_Float16)(accX[nt][r] + bxv);
        }
    }
    // Stage Wslice (64x32) into B-fragment layout: frag (kc, nt) at (kc*2+nt)*512
    {
        int d = tid >> 2, g0 = (tid & 3) << 3;
        const v4f* wsp = (const v4f*)(Wslice + d * GG + g0);
        float tw[8];
        *(v4f*)(tw + 0) = wsp[0]; *(v4f*)(tw + 4) = wsp[1];
        int kc = d >> 5, kr = d & 31;
#pragma unroll
        for (int j = 0; j < 8; ++j) {
            int g = g0 + j;
            sWs[(kc * 2 + (g >> 4)) * 512 + b_swizzle(kr, g & 15)] = (_Float16)tw[j];
        }
    }
    if (tid < 32) snorm[tid] = 0.0f;
    __syncthreads();

    // slice logits: x_mid(128x64) @ Wslice(64x32) via WMMA
    const v16h* pXm = (const v16h*)sXm;
    const v16h* pWs = (const v16h*)sWs;
    v8f accL[2] = {};
#pragma unroll
    for (int kc = 0; kc < 2; ++kc) {
        v16h a = pXm[(wave * 2 + kc) * 32 + lane];
#pragma unroll
        for (int nt = 0; nt < 2; ++nt)
            accL[nt] = WMMA_F16(a, pWs[(kc * 2 + nt) * 32 + lane], accL[nt]);
    }
    float invT = 1.0f / temp[h];
#pragma unroll
    for (int nt = 0; nt < 2; ++nt) {
        int g = nt * 16 + nlo;
        float bs = bslice[g];
#pragma unroll
        for (int r = 0; r < 8; ++r) {
            int tr = wave * 16 + mhi + r;
            sLog[tr * 32 + g] = (accL[nt][r] + bs) * invT;
        }
    }
    __syncthreads();

    // Softmax over G=32 per row; write slice_weights; accumulate norm
    if (tid < 128) {
        int tr = tid;
        int n  = nbase + tr;
        float mx = -3.4e38f;
#pragma unroll
        for (int g = 0; g < 32; ++g) mx = fmaxf(mx, sLog[tr * 32 + g]);
        float e[32];
        float sum = 0.0f;
#pragma unroll
        for (int g = 0; g < 32; ++g) {
            e[g] = __expf(sLog[tr * 32 + g] - mx);
            sum += e[g];
        }
        float inv = 1.0f / sum;
        size_t base = ((size_t)(b * HH + h) * NN + n) * GG;
#pragma unroll
        for (int g = 0; g < 32; ++g) {
            float w = e[g] * inv;
            sw_out[base + g] = w;
            atomicAdd(&snorm[g], w);
        }
    }
    __syncthreads();
    if (tid < 32) atomicAdd(&slice_norm[(b * HH + h) * GG + tid], snorm[tid]);
}

// ---------------------------------------------------------------------------
// K2: split-K soft pooling: partial[bh,split,g,d] = sum_n fx[n,d]*sw[n,g]
// grid = (B*H, 16), block 256; each thread owns (g, 8 d's).
// ---------------------------------------------------------------------------
__global__ void __launch_bounds__(256)
pool_kernel(const _Float16* __restrict__ fx, const float* __restrict__ sw,
            float* __restrict__ partial)
{
    __shared__ alignas(16) _Float16 sfx[32 * 64];
    __shared__ alignas(16) float    ssw[32 * 32];
    const int bh = blockIdx.x, split = blockIdx.y;
    const int tid = threadIdx.x;
    const int g  = tid >> 3;
    const int dv = tid & 7;            // v8h column block
    const int n0 = split * (NN / 16);

    float acc[8] = {};
    const v8h* fv = (const v8h*)sfx;

    for (int c = 0; c < 32; ++c) {
        int nb = n0 + c * 32;
        __syncthreads();
        // 2048 halves: thread t -> row t>>3, 8 halves at (t&7)*8 (idx = t*8)
        ((v8h*)sfx)[tid] =
            *(const v8h*)(fx + ((size_t)bh * NN + nb + (tid >> 3)) * DD + ((tid & 7) << 3));
        // 1024 floats: thread t -> row t>>3, 4 floats at (t&7)*4 (idx = t*4)
        ((v4f*)ssw)[tid] =
            *(const v4f*)(sw + ((size_t)bh * NN + nb + (tid >> 3)) * GG + ((tid & 7) << 2));
        __syncthreads();
#pragma unroll 4
        for (int r = 0; r < 32; ++r) {
            float w = ssw[r * 32 + g];
            v8h f = fv[r * 8 + dv];
#pragma unroll
            for (int j = 0; j < 8; ++j)
                acc[j] += w * (float)f[j];
        }
    }
#pragma unroll
    for (int j = 0; j < 8; ++j)
        partial[((size_t)(bh * 16 + split)) * 2048 + g * 64 + (dv << 3) + j] = acc[j];
}

// ---------------------------------------------------------------------------
// K3: reduce partials -> slice_token, tiny attention over G=32 tokens.
// grid = B*H = 32 blocks, block 256.
// ---------------------------------------------------------------------------
__device__ __forceinline__ void st_proj(const float* __restrict__ Wg,
                                        const float* sST, float* sW, float* sOut,
                                        int tid, int g8, int d0)
{
    __syncthreads();
    for (int idx = tid; idx < 4096; idx += 256) sW[idx] = Wg[idx];
    __syncthreads();
    v4f a0 = {}, a1 = {};
    for (int e = 0; e < 64; ++e) {
        float s = sST[g8 * 64 + e];
        const v4f* wrow = (const v4f*)&sW[e * 64 + d0];
        a0 += s * wrow[0];
        a1 += s * wrow[1];
    }
    *(v4f*)&sOut[g8 * 64 + d0]     = a0;
    *(v4f*)&sOut[g8 * 64 + d0 + 4] = a1;
}

__global__ void __launch_bounds__(256)
attn_kernel(const float* __restrict__ partial, const float* __restrict__ norm,
            const float* __restrict__ Wq, const float* __restrict__ Wk,
            const float* __restrict__ Wv, float* __restrict__ out_slice)
{
    __shared__ alignas(16) float sST[32 * 64];
    __shared__ alignas(16) float sW [64 * 64];
    __shared__ alignas(16) float sQ [32 * 64];
    __shared__ alignas(16) float sK [32 * 64];
    __shared__ alignas(16) float sV [32 * 64];
    __shared__ alignas(16) float sS [32 * 32];
    const int bh = blockIdx.x;
    const int tid = threadIdx.x;
    const int g8  = tid >> 3;            // 0..31
    const int d0  = (tid & 7) << 3;      // 0..56

    // reduce split-K partials, normalize by slice_norm
    for (int idx = tid; idx < 2048; idx += 256) {
        float s = 0.0f;
#pragma unroll
        for (int p = 0; p < 16; ++p)
            s += partial[((size_t)(bh * 16 + p)) * 2048 + idx];
        sST[idx] = s / (norm[bh * GG + (idx >> 6)] + 1e-5f);
    }

    st_proj(Wq, sST, sW, sQ, tid, g8, d0);
    st_proj(Wk, sST, sW, sK, tid, g8, d0);
    st_proj(Wv, sST, sW, sV, tid, g8, d0);
    __syncthreads();
    // scores = q @ k^T / sqrt(D)
    for (int idx = tid; idx < 1024; idx += 256) {
        int g = idx >> 5, k = idx & 31;
        const v4f* qp = (const v4f*)&sQ[g * 64];
        const v4f* kp = (const v4f*)&sK[k * 64];
        v4f acc = {};
#pragma unroll
        for (int e = 0; e < 16; ++e) acc += qp[e] * kp[e];
        sS[idx] = (acc[0] + acc[1] + acc[2] + acc[3]) * 0.125f;
    }
    __syncthreads();
    // row softmax (32 rows)
    if (tid < 32) {
        float mx = -3.4e38f;
#pragma unroll
        for (int k = 0; k < 32; ++k) mx = fmaxf(mx, sS[tid * 32 + k]);
        float sum = 0.0f;
#pragma unroll
        for (int k = 0; k < 32; ++k) {
            float e = __expf(sS[tid * 32 + k] - mx);
            sS[tid * 32 + k] = e;
            sum += e;
        }
        float inv = 1.0f / sum;
#pragma unroll
        for (int k = 0; k < 32; ++k) sS[tid * 32 + k] *= inv;
    }
    __syncthreads();
    // out_slice = attn @ v
    {
        v4f a0 = {}, a1 = {};
#pragma unroll 8
        for (int k = 0; k < 32; ++k) {
            float s = sS[g8 * 32 + k];
            const v4f* vrow = (const v4f*)&sV[k * 64 + d0];
            a0 += s * vrow[0];
            a1 += s * vrow[1];
        }
        *(v4f*)&out_slice[(size_t)bh * 2048 + g8 * 64 + d0]     = a0;
        *(v4f*)&out_slice[(size_t)bh * 2048 + g8 * 64 + d0 + 4] = a1;
    }
}

// ---------------------------------------------------------------------------
// K4: scatter back: out_x[n, h*64+d] = sum_g sw[b,h,n,g] * out_slice[b,h,g,d]
// grid = (N/128, B*H), block 256. Thread owns one row-half (32 d's).
// ---------------------------------------------------------------------------
__global__ void __launch_bounds__(256)
expand_kernel(const float* __restrict__ sw, const float* __restrict__ os,
              _Float16* __restrict__ out_x)
{
    __shared__ alignas(16) float sOS[32 * 64];
    __shared__ alignas(16) float sSW[128 * 32];
    const int n0 = blockIdx.x * 128;
    const int bh = blockIdx.y;
    const int b = bh >> 3, h = bh & 7;
    const int tid = threadIdx.x;

    for (int idx = tid; idx < 2048; idx += 256)
        sOS[idx] = os[(size_t)bh * 2048 + idx];
    // 4096 floats of sw: thread t -> row t>>1, 16 floats
    {
        int r = tid >> 1, g0 = (tid & 1) << 4;
        const v4f* sp = (const v4f*)(sw + ((size_t)bh * NN + n0 + r) * GG + g0);
        v4f* dp = (v4f*)&sSW[r * 32 + g0];
        dp[0] = sp[0]; dp[1] = sp[1]; dp[2] = sp[2]; dp[3] = sp[3];
    }
    __syncthreads();

    const int r  = tid >> 1;
    const int d0 = (tid & 1) << 5;    // 0 or 32
    v4f acc[8] = {};
#pragma unroll 4
    for (int g = 0; g < 32; ++g) {
        float w = sSW[r * 32 + g];
        const v4f* osp = (const v4f*)&sOS[g * 64 + d0];
#pragma unroll
        for (int j = 0; j < 8; ++j) acc[j] += w * osp[j];
    }
    _Float16 tmp[32];
#pragma unroll
    for (int j = 0; j < 8; ++j) {
#pragma unroll
        for (int c = 0; c < 4; ++c) tmp[j * 4 + c] = (_Float16)acc[j][c];
    }
    _Float16* op = out_x + (size_t)(b * NN + n0 + r) * INNER + h * 64 + d0;
#pragma unroll
    for (int j = 0; j < 4; ++j)
        ((v8h*)op)[j] = *(const v8h*)&tmp[j * 8];
}

// ---------------------------------------------------------------------------
// K5: out = out_x(65536x512 f16) @ Wout(512x256) + bout (WMMA)
// grid = (512, 4), block 256 (8 waves); 128x64 tile per block.
// ---------------------------------------------------------------------------
__global__ void __launch_bounds__(256)
out_gemm_kernel(const _Float16* __restrict__ A, const float* __restrict__ Wout,
                const float* __restrict__ bout, float* __restrict__ out)
{
    __shared__ alignas(32) _Float16 sA[8 * 512];
    __shared__ alignas(32) _Float16 sB[4 * 512];
    const int row0 = blockIdx.x * 128;
    const int col0 = blockIdx.y * 64;
    const int tid = threadIdx.x;
    const int wave = tid >> 5;
    const int lane = tid & 31;

    v8f acc[4] = {};
    const v16h* pA = (const v16h*)sA;
    const v16h* pB = (const v16h*)sB;

    for (int ks = 0; ks < INNER; ks += 32) {
        __syncthreads();
        {
            int r = tid >> 1, kk0 = (tid & 1) << 4;
            const v8h* ap = (const v8h*)(A + (size_t)(row0 + r) * INNER + ks + kk0);
            v8h h0 = ap[0], h1 = ap[1];
            int mt = r >> 4, m = r & 15;
#pragma unroll
            for (int j = 0; j < 8; ++j) {
                sA[mt * 512 + a_swizzle(m, kk0 + j)]     = h0[j];
                sA[mt * 512 + a_swizzle(m, kk0 + 8 + j)] = h1[j];
            }
        }
        {
            int kr = tid >> 3, c0 = (tid & 7) << 3;
            const v4f* wp = (const v4f*)(Wout + (size_t)(ks + kr) * DIMM + col0 + c0);
            float tw[8];
            *(v4f*)(tw + 0) = wp[0]; *(v4f*)(tw + 4) = wp[1];
#pragma unroll
            for (int j = 0; j < 8; ++j) {
                int c = c0 + j;
                sB[(c >> 4) * 512 + b_swizzle(kr, c & 15)] = (_Float16)tw[j];
            }
        }
        __syncthreads();
        v16h a = pA[wave * 32 + lane];
#pragma unroll
        for (int nt = 0; nt < 4; ++nt)
            acc[nt] = WMMA_F16(a, pB[nt * 32 + lane], acc[nt]);
    }

    const int nlo = lane & 15;
    const int mhi = (lane >> 4) << 3;
#pragma unroll
    for (int nt = 0; nt < 4; ++nt) {
        int col = col0 + nt * 16 + nlo;
        float bv = bout[col];
#pragma unroll
        for (int r = 0; r < 8; ++r) {
            int row = row0 + wave * 16 + mhi + r;
            out[(size_t)row * DIMM + col] = acc[nt][r] + bv;
        }
    }
}

// ---------------------------------------------------------------------------
// Launch
// ---------------------------------------------------------------------------
extern "C" void kernel_launch(void* const* d_in, const int* in_sizes, int n_in,
                              void* d_out, int out_size, void* d_ws, size_t ws_size,
                              hipStream_t stream) {
    (void)in_sizes; (void)n_in; (void)out_size; (void)ws_size;
    const float* x      = (const float*)d_in[0];
    const float* Wfx    = (const float*)d_in[1];
    const float* bfx    = (const float*)d_in[2];
    const float* Wx     = (const float*)d_in[3];
    const float* bx     = (const float*)d_in[4];
    const float* Wslice = (const float*)d_in[5];
    const float* bslice = (const float*)d_in[6];
    const float* temp   = (const float*)d_in[7];
    const float* Wq     = (const float*)d_in[8];
    const float* Wk     = (const float*)d_in[9];
    const float* Wv     = (const float*)d_in[10];
    const float* Wout   = (const float*)d_in[11];
    const float* bout   = (const float*)d_in[12];

    char* ws = (char*)d_ws;
    _Float16* fx        = (_Float16*)(ws);                     // 64 MB
    float*    sw        = (float*)   (ws + (64ull  << 20));    // 64 MB
    _Float16* out_x     = (_Float16*)(ws + (128ull << 20));    // 64 MB
    float*    partial   = (float*)   (ws + (192ull << 20));    // 4 MB
    float*    out_slice = (float*)   (ws + (196ull << 20));    // 256 KB
    float*    snorm     = (float*)   (ws + (197ull << 20));    // 4 KB

    (void)hipMemsetAsync(snorm, 0, BB * HH * GG * sizeof(float), stream);

    proj_kernel<<<dim3(512, 8), 256, 0, stream>>>(
        x, Wfx, bfx, Wx, bx, Wslice, bslice, temp, fx, sw, snorm);

    pool_kernel<<<dim3(BB * HH, 16), 256, 0, stream>>>(fx, sw, partial);

    attn_kernel<<<dim3(BB * HH), 256, 0, stream>>>(
        partial, snorm, Wq, Wk, Wv, out_slice);

    expand_kernel<<<dim3(NN / 128, BB * HH), 256, 0, stream>>>(
        sw, out_slice, out_x);

    out_gemm_kernel<<<dim3(512, 4), 256, 0, stream>>>(
        out_x, Wout, bout, (float*)d_out);
}
